// Net_23845658428280
// MI455X (gfx1250) — matrix operand order repaired and checked by reading, not compiled, not run on hardware
//
#include <hip/hip_runtime.h>
#include <hip/hip_bf16.h>
#include <math.h>

typedef __attribute__((ext_vector_type(16))) _Float16 v16h;
typedef __attribute__((ext_vector_type(8)))  float    v8f;

#define NB   256      // batch
#define NT   1024     // timesteps
#define IN0  128      // input features, layer 0
#define HID  64       // hidden
#define G4   256      // 4*HID gate width
#define BT   16       // batch rows per workgroup
#define NWG  (NB/BT)  // 16 batch tiles; 2*NWG workgroups total (L0 + L1)

union Frag16 { v16h h; unsigned u[8]; };

static __device__ __forceinline__ unsigned pack2(float a, float b) {
    union { _Float16 h[2]; unsigned u; } p;
    p.h[0] = (_Float16)a; p.h[1] = (_Float16)b;
    return p.u;
}

// ---- branch-free native activations (keep the sequential loop VALU-lean) ----
#if __has_builtin(__builtin_amdgcn_exp2f)
static __device__ __forceinline__ float fexp2(float x) { return __builtin_amdgcn_exp2f(x); }
#else
static __device__ __forceinline__ float fexp2(float x) { return exp2f(x); }
#endif

#if __has_builtin(__builtin_amdgcn_rcpf)
static __device__ __forceinline__ float frcp(float x) { return __builtin_amdgcn_rcpf(x); }
#else
static __device__ __forceinline__ float frcp(float x) { return 1.0f / x; }
#endif

static __device__ __forceinline__ float fsigmoid(float x) {
    return frcp(1.0f + fexp2(-1.44269504f * x));
}

#if __has_builtin(__builtin_amdgcn_tanhf)
static __device__ __forceinline__ float ftanh(float x) { return __builtin_amdgcn_tanhf(x); }
#else
static __device__ __forceinline__ float ftanh(float x) {
    return 1.0f - 2.0f * frcp(1.0f + fexp2(2.88539008f * x));
}
#endif

// ---------------------------------------------------------------------------
// One LSTM layer over a 16-row batch tile, persistent over all T steps.
//   K        : input feature width (128 for L0, 64 for L1)
//   SRC32    : input is f32 [B][T][K] (layer 0 reads raw x, converts on stage)
//   XT       : input is f16 time-major [T][B][K] (pipelined h1 from layer 0)
//   FUSED_OUT: apply w_out/b_out + sigmoid, write d_out; no hidden stored
//   PRODUCE  : publish per-step ticket (release, agent scope) after h write
//   CONSUME  : acquire-spin on producer's ticket before staging step t+1
// hout (PRODUCE path) uses time-major [T][B][HID] so step t and t+1 never
// share a cacheline (consumer WGP$ safety without scoped loads).
// ---------------------------------------------------------------------------
template <int K, bool SRC32, bool XT, bool FUSED_OUT, bool PRODUCE, bool CONSUME>
static __device__ __forceinline__ void lstm_layer_body(
    const void*     __restrict__ xin_v,
    const float*    __restrict__ w_ih,
    const float*    __restrict__ w_hh,
    const float*    __restrict__ b_ih,
    const float*    __restrict__ b_hh,
    _Float16*       __restrict__ hout,
    const float*    __restrict__ w_out,
    const float*    __restrict__ b_out,
    float*          __restrict__ out,
    int*            __restrict__ flag,   // ticket: produced step count
    int             wg)                  // batch-tile index 0..NWG-1
{
    constexpr int KT_IN = K / 32;
    constexpr int KT_H  = 2;
    constexpr int NLOAD = 2 * K;    // threads staging 8 f16 elements each

    __shared__ _Float16 xs[2][BT * K];
    __shared__ _Float16 hs[BT * HID];
    __shared__ __align__(16) float gb[BT * G4];

    const int tid  = threadIdx.x;
    const int wave = tid >> 5;
    const int lane = tid & 31;
    const int b0   = wg * BT;

    const int nrow = lane & 15;
    const int kbB  = (lane >> 4) << 4;
    const int m    = lane & 15;
    const int khbA = (lane >> 4) << 3;

    // ---- weight B-fragments (B[k][n] = W[n][k]) resident in registers
    Frag16 wih[2][KT_IN];
    Frag16 whh[2][KT_H];
    float  bias[2];
#pragma unroll
    for (int nt = 0; nt < 2; ++nt) {
        const int n = wave * 32 + nt * 16 + nrow;
#pragma unroll
        for (int kt = 0; kt < KT_IN; ++kt)
#pragma unroll
            for (int v = 0; v < 8; ++v) {
                const int k = kt * 32 + khbA + ((v & 3) << 1) + ((v >> 2) << 4);
                wih[nt][kt].u[v] = pack2(w_ih[n * K + k], w_ih[n * K + k + 1]);
            }
#pragma unroll
        for (int kt = 0; kt < KT_H; ++kt)
#pragma unroll
            for (int v = 0; v < 8; ++v) {
                const int k = kt * 32 + khbA + ((v & 3) << 1) + ((v >> 2) << 4);
                whh[nt][kt].u[v] = pack2(w_hh[n * HID + k], w_hh[n * HID + k + 1]);
            }
        bias[nt] = b_ih[n] + b_hh[n];
    }

    // ---- per-thread cell state: row = tid/16, cols col0..col0+3
    const int erow = tid >> 4;
    const int col0 = (tid & 15) * 4;
    float creg[4] = {0.f, 0.f, 0.f, 0.f};
#pragma unroll
    for (int j = 0; j < 4; ++j) hs[erow * HID + col0 + j] = (_Float16)0.f;

    float wcol[4];
    float bo = 0.f;
    if constexpr (FUSED_OUT) {
#pragma unroll
        for (int j = 0; j < 4; ++j) wcol[j] = w_out[col0 + j];
        bo = b_out[0];
    }

    // ---- input staging: loader threads each own one 8-element chunk
    const bool loader = (tid < NLOAD);
    const int  lrow   = tid / (K / 8);
    const int  lch    = tid % (K / 8);
    uint4 xreg;

    auto wait_ready = [&](int steps_needed) {
        if constexpr (CONSUME) {
            while (__hip_atomic_load(flag, __ATOMIC_ACQUIRE,
                                     __HIP_MEMORY_SCOPE_AGENT) < steps_needed)
                __builtin_amdgcn_s_sleep(1);
        }
    };

    auto load_x = [&](int t) {
        size_t gidx;
        if constexpr (XT) gidx = ((size_t)t * NB + (b0 + lrow)) * K + lch * 8;
        else              gidx = ((size_t)(b0 + lrow) * NT + t) * K + lch * 8;
        if constexpr (SRC32) {
            const float* xf = (const float*)xin_v;
            const float4 a = *(const float4*)(xf + gidx);
            const float4 b = *(const float4*)(xf + gidx + 4);
            xreg.x = pack2(a.x, a.y); xreg.y = pack2(a.z, a.w);
            xreg.z = pack2(b.x, b.y); xreg.w = pack2(b.z, b.w);
            if (t + 1 < NT)
                __builtin_prefetch(xf + gidx + K, 0, 0);    // global_prefetch_b8
        } else {
            const _Float16* xh = (const _Float16*)xin_v;
            xreg = *(const uint4*)(xh + gidx);
            // no speculative prefetch on the CONSUME path: data may not exist yet
            if constexpr (!CONSUME) {
                if (t + 1 < NT) __builtin_prefetch(xh + gidx + K, 0, 0);
            }
        }
    };

    if (loader) {
        wait_ready(1);
        load_x(0);
        *(uint4*)(&xs[0][lrow * K + lch * 8]) = xreg;
    }
    __syncthreads();

    for (int t = 0; t < NT; ++t) {
        const int cur = t & 1;
        const int nxt = cur ^ 1;

        // stage t+1 early; consumer first waits for producer's ticket
        if (loader && (t + 1 < NT)) {
            wait_ready(t + 2);
            load_x(t + 1);
        }

        // A-fragments (ISA 16-bit 16x32 A layout)
        Frag16 xa[KT_IN], ha[KT_H];
#pragma unroll
        for (int kt = 0; kt < KT_IN; ++kt)
#pragma unroll
            for (int v = 0; v < 8; ++v) {
                const int k = kt * 32 + khbA + ((v & 3) << 1) + ((v >> 2) << 4);
                xa[kt].u[v] = *(const unsigned*)&xs[cur][m * K + k];
            }
#pragma unroll
        for (int kt = 0; kt < KT_H; ++kt)
#pragma unroll
            for (int v = 0; v < 8; ++v) {
                const int k = kt * 32 + khbA + ((v & 3) << 1) + ((v >> 2) << 4);
                ha[kt].u[v] = *(const unsigned*)&hs[m * HID + k];
            }

        // WMMA with dual accumulators (half the dependency depth)
#pragma unroll
        for (int nt = 0; nt < 2; ++nt) {
            v8f acc0, acc1;
#pragma unroll
            for (int r = 0; r < 8; ++r) { acc0[r] = bias[nt]; acc1[r] = 0.f; }
#pragma unroll
            for (int kt = 0; kt < KT_IN; ++kt) {
                if (kt & 1)
                    acc1 = __builtin_amdgcn_wmma_f32_16x16x32_f16(
                            false, xa[kt].h, false, wih[nt][kt].h,
                            (short)0, acc1, false, false);
                else
                    acc0 = __builtin_amdgcn_wmma_f32_16x16x32_f16(
                            false, xa[kt].h, false, wih[nt][kt].h,
                            (short)0, acc0, false, false);
            }
#pragma unroll
            for (int kt = 0; kt < KT_H; ++kt) {
                if (kt & 1)
                    acc1 = __builtin_amdgcn_wmma_f32_16x16x32_f16(
                            false, ha[kt].h, false, whh[nt][kt].h,
                            (short)0, acc1, false, false);
                else
                    acc0 = __builtin_amdgcn_wmma_f32_16x16x32_f16(
                            false, ha[kt].h, false, whh[nt][kt].h,
                            (short)0, acc0, false, false);
            }
            const int col = wave * 32 + nt * 16 + nrow;
            const int rb  = (lane >> 4) << 3;
#pragma unroll
            for (int r = 0; r < 8; ++r)
                gb[(rb + r) * G4 + col] = acc0[r] + acc1[r];
        }

        if (loader && (t + 1 < NT))
            *(uint4*)(&xs[nxt][lrow * K + lch * 8]) = xreg;

        __syncthreads();   // gates complete, x[t+1] staged

        // cell update
        const float4 gi4 = *(const float4*)&gb[erow * G4 +           col0];
        const float4 gf4 = *(const float4*)&gb[erow * G4 +  HID    + col0];
        const float4 gg4 = *(const float4*)&gb[erow * G4 + 2 * HID + col0];
        const float4 go4 = *(const float4*)&gb[erow * G4 + 3 * HID + col0];
        const float gi[4] = {gi4.x, gi4.y, gi4.z, gi4.w};
        const float gf[4] = {gf4.x, gf4.y, gf4.z, gf4.w};
        const float gg[4] = {gg4.x, gg4.y, gg4.z, gg4.w};
        const float go[4] = {go4.x, go4.y, go4.z, go4.w};

        union { _Float16 h[4]; uint2 u2; } hv;
        float part = 0.f;
#pragma unroll
        for (int j = 0; j < 4; ++j) {
            const float ig = fsigmoid(gi[j]);
            const float fg = fsigmoid(gf[j]);
            const float og = fsigmoid(go[j]);
            const float cc = fg * creg[j] + ig * ftanh(gg[j]);
            creg[j] = cc;
            const float hh = og * ftanh(cc);
            hs[erow * HID + col0 + j] = (_Float16)hh;
            hv.h[j] = (_Float16)hh;
            if constexpr (FUSED_OUT) part += hh * wcol[j];
        }
        if constexpr (FUSED_OUT) {
            part += __shfl_xor(part, 1);
            part += __shfl_xor(part, 2);
            part += __shfl_xor(part, 4);
            part += __shfl_xor(part, 8);
            if ((lane & 15) == 0)
                out[(size_t)(b0 + erow) * NT + t] = fsigmoid(part + bo);
        } else {
            // time-major h1: [T][B][HID]
            *(uint2*)(hout + ((size_t)t * NB + (b0 + erow)) * HID + col0) = hv.u2;
        }

        if constexpr (PRODUCE) __threadfence();   // h tile visible agent-wide
        __syncthreads();                          // hs ready for next WMMA
        if constexpr (PRODUCE) {
            if (tid == 0)
                __hip_atomic_store(flag, t + 1, __ATOMIC_RELEASE,
                                   __HIP_MEMORY_SCOPE_AGENT);
        }
    }
}

// ---------------------------------------------------------------------------
// Fused 2-layer pipelined kernel: WGs 0..NWG-1 = layer 0 (producer),
// WGs NWG..2*NWG-1 = layer 1 (consumer + fused sigmoid head).
// ---------------------------------------------------------------------------
__global__ __launch_bounds__(256) void lstm2_pipe_kernel(
    const float* __restrict__ x,
    const float* __restrict__ w_ih0, const float* __restrict__ w_hh0,
    const float* __restrict__ b_ih0, const float* __restrict__ b_hh0,
    const float* __restrict__ w_ih1, const float* __restrict__ w_hh1,
    const float* __restrict__ b_ih1, const float* __restrict__ b_hh1,
    const float* __restrict__ w_out, const float* __restrict__ b_out,
    _Float16*    __restrict__ h1,    // [NT][NB][HID] f16, time-major
    int*         __restrict__ flags, // [NWG] produced-step tickets
    float*       __restrict__ out)   // [NB][NT] f32
{
    const int bid = blockIdx.x;
    if (bid < NWG) {
        // layer 0: f32 x in, produce time-major f16 h1 + tickets
        lstm_layer_body<IN0, true, false, false, true, false>(
            x, w_ih0, w_hh0, b_ih0, b_hh0, h1,
            nullptr, nullptr, nullptr, flags + bid, bid);
    } else {
        // layer 1: consume h1 behind tickets, fused sigmoid head to d_out
        const int wg = bid - NWG;
        lstm_layer_body<HID, false, true, true, false, true>(
            h1, w_ih1, w_hh1, b_ih1, b_hh1, nullptr,
            w_out, b_out, out, flags + wg, wg);
    }
}

__global__ void reset_flags_kernel(int* flags) {
    if (threadIdx.x < NWG) flags[threadIdx.x] = 0;
}

// ---------------------------------------------------------------------------
extern "C" void kernel_launch(void* const* d_in, const int* in_sizes, int n_in,
                              void* d_out, int out_size, void* d_ws, size_t ws_size,
                              hipStream_t stream)
{
    const float* x     = (const float*)d_in[0];
    const float* w_ih0 = (const float*)d_in[1];
    const float* w_hh0 = (const float*)d_in[2];
    const float* b_ih0 = (const float*)d_in[3];
    const float* b_hh0 = (const float*)d_in[4];
    const float* w_ih1 = (const float*)d_in[5];
    const float* w_hh1 = (const float*)d_in[6];
    const float* b_ih1 = (const float*)d_in[7];
    const float* b_hh1 = (const float*)d_in[8];
    const float* w_out = (const float*)d_in[9];
    const float* b_out = (const float*)d_in[10];
    float* out = (float*)d_out;

    // workspace: h1 [NT*NB*HID] f16 (time-major) | flags [NWG] int
    _Float16* h1    = (_Float16*)d_ws;
    int*      flags = (int*)((char*)d_ws + (size_t)NT * NB * HID * sizeof(_Float16));

    reset_flags_kernel<<<1, 64, 0, stream>>>(flags);

    lstm2_pipe_kernel<<<2 * NWG, 256, 0, stream>>>(
        x, w_ih0, w_hh0, b_ih0, b_hh0,
        w_ih1, w_hh1, b_ih1, b_hh1,
        w_out, b_out, h1, flags, out);
}
